// SequenceLayer_6975026889282
// MI455X (gfx1250) — compile-verified, matrix operand order checked by reading, not run on hardware
//
#include <hip/hip_runtime.h>
#include <math.h>

typedef __attribute__((ext_vector_type(16))) __bf16 v16bf;
typedef __attribute__((ext_vector_type(8)))  float  v8f;
typedef __attribute__((ext_vector_type(4)))  unsigned int u32x4;
typedef __attribute__((ext_vector_type(8)))  unsigned int u32x8;

#define TOK   32768   // B*L
#define HDIM  512
#define PDIM  512
#define LSEQ  4096
#define NBATCH 8
#define NCHUNK 16
#define CHUNK  256

// ---------------------------------------------------------------------------
// Prep: Lambda_bar = bl*(1+step/2*Lam), bl = 1/(1-step/2*Lam)
// ---------------------------------------------------------------------------
__device__ __forceinline__ void c_recip(float dr, float di, float& rr, float& ri) {
    float d = dr * dr + di * di;
    rr = dr / d; ri = -di / d;
}

__global__ __launch_bounds__(256) void k_params(
    const float* __restrict__ lre_, const float* __restrict__ lim_,
    const float* __restrict__ log_step, const float* __restrict__ rate,
    float* __restrict__ lamb) {
    int p = blockIdx.x * 256 + threadIdx.x;
    if (p >= PDIM) return;
    float step = rate[0] * __expf(log_step[p]);
    float hs = 0.5f * step;
    float lre = lre_[p], lim = lim_[p];
    float br, bi; c_recip(1.f - hs * lre, -hs * lim, br, bi);
    float nr = 1.f + hs * lre, ni = hs * lim;
    lamb[2 * p]     = br * nr - bi * ni;
    lamb[2 * p + 1] = br * ni + bi * nr;
}

// Pack B_bar = (bl*step)*(B_re + i B_im) as bf16 rows [re(512); im(512)] x H
__global__ __launch_bounds__(256) void k_bbar(
    const float* __restrict__ lre_, const float* __restrict__ lim_,
    const float* __restrict__ log_step, const float* __restrict__ rate,
    const float* __restrict__ Bre, const float* __restrict__ Bim,
    __bf16* __restrict__ Wb) {
    int i = blockIdx.x * 256 + threadIdx.x;
    if (i >= PDIM * HDIM) return;
    int p = i >> 9, h = i & 511;
    float step = rate[0] * __expf(log_step[p]);
    float hs = 0.5f * step;
    float br, bi; c_recip(1.f - hs * lre_[p], -hs * lim_[p], br, bi);
    float bsr = br * step, bsi = bi * step;
    float xr = Bre[i], xi = Bim[i];
    Wb[(size_t)p * HDIM + h]            = (__bf16)(bsr * xr - bsi * xi);
    Wb[(size_t)(PDIM + p) * HDIM + h]   = (__bf16)(bsr * xi + bsi * xr);
}

// Pack [C_re | -C_im] as (H x 2P) bf16 for Re(C*xs) via one K=1024 GEMM
__global__ __launch_bounds__(256) void k_w2(
    const float* __restrict__ Cre, const float* __restrict__ Cim,
    __bf16* __restrict__ W2) {
    int i = blockIdx.x * 256 + threadIdx.x;
    if (i >= HDIM * PDIM) return;
    int h = i >> 9, p = i & 511;
    W2[(size_t)h * 1024 + p]        = (__bf16)Cre[i];
    W2[(size_t)h * 1024 + 512 + p]  = (__bf16)(-Cim[i]);
}

__global__ __launch_bounds__(256) void k_w3(
    const float* __restrict__ W, __bf16* __restrict__ Wo) {
    int i = blockIdx.x * 256 + threadIdx.x;
    if (i >= HDIM * HDIM) return;
    Wo[i] = (__bf16)W[i];
}

// ---------------------------------------------------------------------------
// LayerNorm: one wave per token (wave32), 16 elems/lane, shfl reductions
// ---------------------------------------------------------------------------
__global__ __launch_bounds__(256) void k_layernorm(
    const float* __restrict__ x, const float* __restrict__ gamma,
    const float* __restrict__ beta, __bf16* __restrict__ xn) {
    int token = blockIdx.x * 8 + (threadIdx.x >> 5);
    int lane  = threadIdx.x & 31;
    const float* row = x + (size_t)token * HDIM;
    float v[16], s = 0.f, s2 = 0.f;
#pragma unroll
    for (int j = 0; j < 16; ++j) {
        v[j] = row[lane + 32 * j];
        s += v[j]; s2 += v[j] * v[j];
    }
#pragma unroll
    for (int off = 16; off >= 1; off >>= 1) {
        s  += __shfl_xor(s, off, 32);
        s2 += __shfl_xor(s2, off, 32);
    }
    float mu   = s * (1.f / HDIM);
    float var  = s2 * (1.f / HDIM) - mu * mu;
    float rstd = rsqrtf(var + 1e-5f);
    __bf16* orow = xn + (size_t)token * HDIM;
#pragma unroll
    for (int j = 0; j < 16; ++j) {
        int h = lane + 32 * j;
        orow[h] = (__bf16)((v[j] - mu) * rstd * gamma[h] + beta[h]);
    }
}

// ---------------------------------------------------------------------------
// TDM: DMA a (rows x 32) bf16 tile (row stride K elements) from global memory
// into LDS, padding each 64B row to 80B (pad_interval=16 DWORDs, pad_amount=4
// DWORDs) so fragment ds_load_b128 reads are bank-conflict-free.
// D# per CDNA5 ISA: group0 = {flags, lds_addr, gaddr_lo,
// gaddr_hi|type=2}; group1 = dims/strides. Groups 2/3 omitted (2-D tensor).
// Issued by one wave; tracked on TENSORcnt.
// ---------------------------------------------------------------------------
__device__ __forceinline__ void tdm_load_tile(const __bf16* gptr, unsigned lds_addr,
                                              int K, int rows) {
    unsigned long long ga = (unsigned long long)(uintptr_t)gptr;
    u32x4 g0;
    g0[0] = 1u;                                           // count=1, user mode
    g0[1] = lds_addr;                                     // LDS byte address
    g0[2] = (unsigned)(ga & 0xFFFFFFFFu);                 // global_addr[31:0]
    g0[3] = (unsigned)((ga >> 32) & 0x1FFFFFFu) | (2u << 30); // [56:32] | type=2
    u32x8 g1;
    g1[0] = (1u << 16)        // data_size = 1 (2 bytes)
          | (1u << 20)        // pad_enable
          | (3u << 22)        // pad_interval = 16 DWORDs (one 64B row)
          | (3u << 25);       // pad_amount   = 4 DWORDs (16B)
    g1[1] = ((unsigned)K & 0xFFFFu) << 16;                     // tensor_dim0[15:0]
    g1[2] = ((unsigned)K >> 16) | (((unsigned)rows) << 16);    // dim0 hi | tensor_dim1 lo
    g1[3] = (32u << 16);                                       // dim1 hi=0 | tile_dim0=32
    g1[4] = (unsigned)rows;                                    // tile_dim1 | tile_dim2=0
    g1[5] = (unsigned)K;                                       // tensor_dim0_stride[31:0]
    g1[6] = 0u;                                                // stride hi | dim1_stride lo
    g1[7] = 0u;
    asm volatile("tensor_load_to_lds %0, %1" :: "s"(g0), "s"(g1) : "memory");
}

// ---------------------------------------------------------------------------
// WMMA GEMM: C(M,N) = A(MxK bf16) * W(NxK bf16)^T, fp32 accumulate.
// Block tile 128x128, 8 waves, wave tile 32x64 = 2x4 frags of 16x16.
// Double-buffered LDS tiles filled by the Tensor Data Mover: wave 0 issues the
// DMA for tile k+1 while all waves run WMMAs on tile k.
// mode 0: store fp32
// mode 1: val = acc + D[n]*xn[m,n]; y=val (fp32); x1 = gelu(val) (bf16)
// mode 2: out[m,n] = skip + yin * sigmoid(acc + bias[n])
// ---------------------------------------------------------------------------
__global__ __launch_bounds__(256) void k_gemm(
    const __bf16* __restrict__ A, const __bf16* __restrict__ W,
    int M, int N, int K, int mode,
    float* __restrict__ C0,
    const float* __restrict__ e0,   // mode1: D        mode2: bias
    const __bf16* __restrict__ e1,  // mode1: xn bf16
    __bf16* __restrict__ e2,        // mode1: x1 out
    const float* __restrict__ e3,   // mode2: y in
    const float* __restrict__ e4) { // mode2: skip
    __shared__ __bf16 sA[2][128 * 40];
    __shared__ __bf16 sW[2][128 * 40];
    int t  = threadIdx.x;
    int m0 = blockIdx.y * 128;
    int n0 = blockIdx.x * 128;
    int wv = t >> 5, lane = t & 31;
    int mb = (wv & 3) * 32;     // wave M offset within block tile
    int nb = (wv >> 2) * 64;    // wave N offset
    int lm = lane & 15, hi = lane >> 4;

    unsigned ldsA[2], ldsW[2];
    ldsA[0] = (unsigned)(uintptr_t)&sA[0][0];
    ldsA[1] = (unsigned)(uintptr_t)&sA[1][0];
    ldsW[0] = (unsigned)(uintptr_t)&sW[0][0];
    ldsW[1] = (unsigned)(uintptr_t)&sW[1][0];

    v8f acc[2][4];
#pragma unroll
    for (int i = 0; i < 2; ++i)
#pragma unroll
        for (int j = 0; j < 4; ++j)
#pragma unroll
            for (int r = 0; r < 8; ++r) acc[i][j][r] = 0.f;

    int nk = K >> 5;
    if (t < 32) {   // wave 0: prime the pipeline with tile pair 0
        tdm_load_tile(A + (size_t)m0 * K, ldsA[0], K, 128);
        tdm_load_tile(W + (size_t)n0 * K, ldsW[0], K, 128);
    }

    for (int kk = 0; kk < nk; ++kk) {
        int cur = kk & 1;
        __syncthreads();   // all waves done reading buf[1-cur] before DMA refill
        if (t < 32) {
            if (kk + 1 < nk) {
                tdm_load_tile(A + (size_t)m0 * K + (kk + 1) * 32, ldsA[1 - cur], K, 128);
                tdm_load_tile(W + (size_t)n0 * K + (kk + 1) * 32, ldsW[1 - cur], K, 128);
                __builtin_amdgcn_s_wait_tensorcnt(2);  // pair kk landed; kk+1 in flight
            } else {
                __builtin_amdgcn_s_wait_tensorcnt(0);
            }
        }
        __syncthreads();   // publish DMA'd tile kk to all waves

        // Fragment packing per ISA 16-bit A layout:
        // lanes 0-15: row=lane, K 0..7 & 16..23 ; lanes 16-31: K 8..15 & 24..31
        v16bf af[2], bfv[4];
#pragma unroll
        for (int i = 0; i < 2; ++i) {
            const __bf16* p = &sA[cur][(mb + i * 16 + lm) * 40 + hi * 8];
            ((uint4*)&af[i])[0] = *(const uint4*)p;
            ((uint4*)&af[i])[1] = *(const uint4*)(p + 16);
        }
#pragma unroll
        for (int j = 0; j < 4; ++j) {
            const __bf16* p = &sW[cur][(nb + j * 16 + lm) * 40 + hi * 8];
            ((uint4*)&bfv[j])[0] = *(const uint4*)p;
            ((uint4*)&bfv[j])[1] = *(const uint4*)(p + 16);
        }
#pragma unroll
        for (int i = 0; i < 2; ++i)
#pragma unroll
            for (int j = 0; j < 4; ++j)
                acc[i][j] = __builtin_amdgcn_wmma_f32_16x16x32_bf16(
                    false, af[i], false, bfv[j], (short)0, acc[i][j], false, false);
    }

    // C/D layout: lanes 0-15: N=lane, M=vgpr r ; lanes 16-31: M=8+r, N=lane-16
#pragma unroll
    for (int i = 0; i < 2; ++i)
#pragma unroll
        for (int j = 0; j < 4; ++j)
#pragma unroll
            for (int r = 0; r < 8; ++r) {
                int m = m0 + mb + i * 16 + hi * 8 + r;
                int n = n0 + nb + j * 16 + lm;
                float v = acc[i][j][r];
                size_t idx = (size_t)m * N + n;
                if (mode == 0) {
                    C0[idx] = v;
                } else if (mode == 1) {
                    float val = v + e0[n] * (float)e1[idx];
                    C0[idx] = val;
                    float gl = 0.5f * val * (1.f + erff(val * 0.7071067811865475f));
                    e2[idx] = (__bf16)gl;
                } else {
                    float gate = 1.f / (1.f + __expf(-(v + e0[n])));
                    C0[idx] = e4[idx] + e3[idx] * gate;
                }
            }
}

// ---------------------------------------------------------------------------
// Chunked parallel scan: x_t = Lam*x_{t-1} + u_t (complex), L split into 16x256
// ---------------------------------------------------------------------------
__global__ __launch_bounds__(256) void k_scan_local(
    float* __restrict__ bu,            // (TOK,1024) [re|im], local scan in-place
    const float* __restrict__ lamb,    // (P,2)
    const float* __restrict__ state,   // (B,P) real
    float* __restrict__ apow, float* __restrict__ xend) {
    int gid = blockIdx.x * 256 + threadIdx.x;      // 65536 threads
    int p = gid & 511;
    int c = (gid >> 9) & 15;
    int b = gid >> 13;
    float lre = lamb[2 * p], lim = lamb[2 * p + 1];
    float xr = 0.f, xi = 0.f, ar = 1.f, ai = 0.f;
    size_t base = ((size_t)b * LSEQ + (size_t)c * CHUNK) * 1024 + p;
    for (int tt = 0; tt < CHUNK; ++tt) {
        float ur = bu[base], ui = bu[base + 512];
        if (c == 0 && tt == 0) {
            float s = state[b * 512 + p];
            ur += lre * s; ui += lim * s;          // + Lambda_bar * state
        }
        float nxr = lre * xr - lim * xi + ur;
        float nxi = lre * xi + lim * xr + ui;
        xr = nxr; xi = nxi;
        bu[base] = xr; bu[base + 512] = xi;
        float nar = ar * lre - ai * lim;
        float nai = ar * lim + ai * lre;
        ar = nar; ai = nai;
        base += 1024;
    }
    int sidx = ((b * 512 + p) * NCHUNK + c) * 2;
    apow[sidx] = ar; apow[sidx + 1] = ai;
    xend[sidx] = xr; xend[sidx + 1] = xi;
}

__global__ __launch_bounds__(256) void k_scan_carry(
    const float* __restrict__ apow, const float* __restrict__ xend,
    float* __restrict__ carry, float* __restrict__ out_state) {
    int gid = blockIdx.x * 256 + threadIdx.x;      // 4096 threads = (b,p)
    float cr = 0.f, ci = 0.f;
    int base = gid * NCHUNK * 2;
    for (int c = 0; c < NCHUNK; ++c) {
        carry[base + 2 * c] = cr; carry[base + 2 * c + 1] = ci;
        float ar = apow[base + 2 * c], ai = apow[base + 2 * c + 1];
        float nr = ar * cr - ai * ci + xend[base + 2 * c];
        float ni = ar * ci + ai * cr + xend[base + 2 * c + 1];
        cr = nr; ci = ni;
    }
    out_state[2 * gid] = cr; out_state[2 * gid + 1] = ci;   // new_state complex
}

__global__ __launch_bounds__(256) void k_scan_fix(
    const float* __restrict__ bu, const float* __restrict__ lamb,
    const float* __restrict__ carry, __bf16* __restrict__ xsb) {
    int gid = blockIdx.x * 256 + threadIdx.x;
    int p = gid & 511;
    int c = (gid >> 9) & 15;
    int b = gid >> 13;
    float lre = lamb[2 * p], lim = lamb[2 * p + 1];
    int cidx = ((b * 512 + p) * NCHUNK + c) * 2;
    float cr = carry[cidx], ci = carry[cidx + 1];
    float ar = lre, ai = lim;                       // Lam^(t+1), t=0
    size_t base = ((size_t)b * LSEQ + (size_t)c * CHUNK) * 1024 + p;
    for (int tt = 0; tt < CHUNK; ++tt) {
        float lr = bu[base], li = bu[base + 512];
        float fr = lr + ar * cr - ai * ci;
        float fi = li + ar * ci + ai * cr;
        xsb[base]       = (__bf16)fr;
        xsb[base + 512] = (__bf16)fi;
        float nar = ar * lre - ai * lim;
        float nai = ar * lim + ai * lre;
        ar = nar; ai = nai;
        base += 1024;
    }
}

// ---------------------------------------------------------------------------
extern "C" void kernel_launch(void* const* d_in, const int* in_sizes, int n_in,
                              void* d_out, int out_size, void* d_ws, size_t ws_size,
                              hipStream_t stream) {
    const float* x     = (const float*)d_in[0];
    const float* state = (const float*)d_in[1];
    const float* rate  = (const float*)d_in[2];
    const float* Lre   = (const float*)d_in[3];
    const float* Lim   = (const float*)d_in[4];
    const float* Bre   = (const float*)d_in[5];
    const float* Bim   = (const float*)d_in[6];
    const float* Cre   = (const float*)d_in[7];
    const float* Cim   = (const float*)d_in[8];
    const float* Dv    = (const float*)d_in[9];
    const float* lstep = (const float*)d_in[10];
    const float* gam   = (const float*)d_in[11];
    const float* bet   = (const float*)d_in[12];
    const float* W3i   = (const float*)d_in[13];
    const float* b2    = (const float*)d_in[14];
    float* out = (float*)d_out;

    char* ws = (char*)d_ws;
    size_t o = 0;
    float*  lamb = (float*)(ws + o);  o += (size_t)2 * PDIM * 4;            // 4 KB
    o = (o + 255) & ~(size_t)255;
    __bf16* Wb   = (__bf16*)(ws + o); o += (size_t)2 * PDIM * HDIM * 2;     // 1 MB
    __bf16* W2   = (__bf16*)(ws + o); o += (size_t)HDIM * 2 * PDIM * 2;     // 1 MB
    __bf16* W3   = (__bf16*)(ws + o); o += (size_t)HDIM * HDIM * 2;         // 0.5 MB
    __bf16* xn   = (__bf16*)(ws + o); o += (size_t)TOK * HDIM * 2;          // 32 MB
    float*  bu   = (float*)(ws + o);  o += (size_t)TOK * 1024 * 4;          // 128 MB
    __bf16* xsb  = (__bf16*)(ws + o); o += (size_t)TOK * 1024 * 2;          // 64 MB
    float*  yb   = (float*)(ws + o);  o += (size_t)TOK * HDIM * 4;          // 64 MB
    __bf16* x1b  = (__bf16*)(ws + o); o += (size_t)TOK * HDIM * 2;          // 32 MB
    float*  apw  = (float*)(ws + o);  o += (size_t)4096 * NCHUNK * 2 * 4;   // 512 KB
    float*  xnd  = (float*)(ws + o);  o += (size_t)4096 * NCHUNK * 2 * 4;
    float*  car  = (float*)(ws + o);  o += (size_t)4096 * NCHUNK * 2 * 4;

    // 1) SSM parameter folding (bf16 weight packs)
    k_params<<<2, 256, 0, stream>>>(Lre, Lim, lstep, rate, lamb);
    k_bbar  <<<1024, 256, 0, stream>>>(Lre, Lim, lstep, rate, Bre, Bim, Wb);
    k_w2    <<<1024, 256, 0, stream>>>(Cre, Cim, W2);
    k_w3    <<<1024, 256, 0, stream>>>(W3i, W3);

    // 2) LayerNorm -> bf16 xn
    k_layernorm<<<TOK / 8, 256, 0, stream>>>(x, gam, bet, xn);

    // 3) Bu = xn @ [B_re;B_im]^T  (M=32768, N=1024, K=512)
    dim3 g1(1024 / 128, TOK / 128);
    k_gemm<<<g1, 256, 0, stream>>>(xn, Wb, TOK, 1024, HDIM, 0, bu,
                                   nullptr, nullptr, nullptr, nullptr, nullptr);

    // 4) Parallel chunked scan along L
    k_scan_local<<<256, 256, 0, stream>>>(bu, lamb, state, apw, xnd);
    k_scan_carry<<<16, 256, 0, stream>>>(apw, xnd, car, out + (size_t)TOK * HDIM);
    k_scan_fix  <<<256, 256, 0, stream>>>(bu, lamb, car, xsb);

    // 5) y = xs @ [C_re|-C_im]^T + D*xn ; x1 = gelu(y)  (K=1024)
    dim3 g2(HDIM / 128, TOK / 128);
    k_gemm<<<g2, 256, 0, stream>>>(xsb, W2, TOK, HDIM, 1024, 1, yb,
                                   Dv, xn, x1b, nullptr, nullptr);

    // 6) out = skip + y * sigmoid(x1 @ W^T + b)
    k_gemm<<<g2, 256, 0, stream>>>(x1b, W3, TOK, HDIM, HDIM, 2, out,
                                   b2, nullptr, nullptr, yb, x);
}
